// CRF_RNN_14499809591875
// MI455X (gfx1250) — matrix operand order repaired and checked by reading, not compiled, not run on hardware
//
#include <hip/hip_runtime.h>

// CRF-RNN mean-field on MI455X (gfx1250, wave32, WMMA f16 16x16x32).
//
// ws layout: [ D_f16: 4*4096*4096*2 = 128 MiB ][ Q_f16: 4*32*4096*2 = 1 MiB ]
// D_f16 fits the 192 MB L2 -> iterations 2..5 stream D from L2, not HBM.
// Q (per batch: 256 KB) is async-copied into LDS once per block and feeds the
// WMMA B operand via ds_load_b128 (bank-conflict-free padded pitch).

typedef __attribute__((ext_vector_type(16))) _Float16 v16h;
typedef __attribute__((ext_vector_type(8)))  _Float16 v8h;
typedef __attribute__((ext_vector_type(4)))  _Float16 v4h;
typedef __attribute__((ext_vector_type(8)))  float    v8f;

#define BS    4
#define NCLS  21
#define NPAD  32
#define NPIX  4096
#define NITER 5

#define QROW       (NPIX + 8)            // LDS row pitch in halves (bank spread)
#define QLDS_BYTES (NPAD * QROW * 2)     // 262,656 B dynamic LDS

// ---------------------------------------------------------------------------
// K1: D[row,:] = W[row,:] / sum(W[row,:]), converted to f16.
// One 256-thread block per row (bs*n = 16384 rows). Row = 16 KB fp32 read,
// 8 KB f16 write. Values kept in registers between the sum and the store.
// ---------------------------------------------------------------------------
__global__ __launch_bounds__(256) void k_rownorm(const float* __restrict__ W,
                                                 _Float16* __restrict__ Dh) {
  __shared__ float red[256];
  const int    tid = threadIdx.x;
  const size_t row = blockIdx.x;                    // 0 .. BS*NPIX-1
  const float4* wrow = (const float4*)(W + row * (size_t)NPIX);

  float4 v[4];
  float  s = 0.f;
#pragma unroll
  for (int j = 0; j < 4; ++j) {
    v[j] = wrow[tid + j * 256];
    s += v[j].x + v[j].y + v[j].z + v[j].w;
  }
  red[tid] = s;
  __syncthreads();
  for (int off = 128; off > 0; off >>= 1) {
    if (tid < off) red[tid] += red[tid + off];
    __syncthreads();
  }
  const float inv = 1.0f / red[0];

  _Float16* drow = Dh + row * (size_t)NPIX;
#pragma unroll
  for (int j = 0; j < 4; ++j) {
    v4h h;
    h[0] = (_Float16)(v[j].x * inv);
    h[1] = (_Float16)(v[j].y * inv);
    h[2] = (_Float16)(v[j].z * inv);
    h[3] = (_Float16)(v[j].w * inv);
    *(v4h*)(drow + (size_t)(tid + j * 256) * 4) = h;
  }
}

// ---------------------------------------------------------------------------
// K2: Q = softmax over the 21-class dim, stored f16 as [b][32][4096]
// (classes 21..31 zero-filled so the padded WMMA columns are exactly 0).
// ---------------------------------------------------------------------------
__global__ __launch_bounds__(256) void k_softmax(const float* __restrict__ q,
                                                 _Float16* __restrict__ Qh) {
  const int g   = blockIdx.x * 256 + threadIdx.x;   // 0 .. BS*NPIX-1
  const int b   = g >> 12;
  const int pix = g & (NPIX - 1);

  const float* qb = q + (size_t)b * NCLS * NPIX + pix;
  float vals[NCLS];
  float mx = -3.0e38f;
#pragma unroll
  for (int d = 0; d < NCLS; ++d) {
    vals[d] = qb[(size_t)d * NPIX];
    mx = fmaxf(mx, vals[d]);
  }
  float sum = 0.f;
#pragma unroll
  for (int d = 0; d < NCLS; ++d) {
    vals[d] = __expf(vals[d] - mx);
    sum += vals[d];
  }
  const float inv = 1.0f / sum;

  _Float16* Qb = Qh + (size_t)b * NPAD * NPIX + pix;
#pragma unroll
  for (int d = 0; d < NCLS; ++d) Qb[(size_t)d * NPIX] = (_Float16)(vals[d] * inv);
#pragma unroll
  for (int d = NCLS; d < NPAD; ++d) Qb[(size_t)d * NPIX] = (_Float16)0.f;
}

// ---------------------------------------------------------------------------
// K3: fused  C[m,e] = sum_n D[m,n]*Q[e,n]   (WMMA f32_16x16x32_f16, e=0..31)
//            out[d,m] = seg[d,m] - sum_e weights[d,e]*C[m,e]
// Each wave owns one 16-pixel tile (A = 16 rows of D streamed from global),
// B (shared by all waves) comes from the LDS-resident Q copy.
// ---------------------------------------------------------------------------
__global__ __launch_bounds__(256, 1) void k_crf_step(
    const _Float16* __restrict__ Dh, const _Float16* __restrict__ Qh,
    const float* __restrict__ seg, const float* __restrict__ wts,
    float* __restrict__ out) {
  extern __shared__ _Float16 Qlds[];     // NPAD * QROW halves (dynamic)
  __shared__ float Cs[8][16][NPAD];
  __shared__ float Wsm[NCLS * NCLS];

  const int tid  = threadIdx.x;
  const int lane = tid & 31;
  const int wave = tid >> 5;

  for (int i = tid; i < NCLS * NCLS; i += 256) Wsm[i] = wts[i];

  const int b    = blockIdx.x >> 5;                    // 32 blocks per batch
  const int tile = ((blockIdx.x & 31) << 3) + wave;    // 0 .. 255
  const int m0   = tile << 4;

  // ---- async preload of this batch's Q (32 x 4096 f16) into LDS ----------
  // 16B chunks: 512 per row, 16384 total; thread t copies chunks t, t+256, ...
  // Each global_load_async_to_lds_b128 moves 512 B per wave (ASYNCcnt).
  {
    const _Float16* Qg = Qh + (size_t)b * NPAD * NPIX;
    for (int ci = tid; ci < NPAD * (NPIX / 8); ci += 256) {
      const int r     = ci >> 9;             // class row
      const int off16 = ci & 511;            // 16B chunk within row
      unsigned long long gaddr =
          (unsigned long long)(uintptr_t)(Qg + (size_t)r * NPIX + off16 * 8);
      unsigned ldst = (unsigned)(uintptr_t)(Qlds + r * QROW + off16 * 8);
      asm volatile("global_load_async_to_lds_b128 %0, %1, off"
                   :: "v"(ldst), "v"(gaddr) : "memory");
    }
    asm volatile("s_wait_asynccnt 0x0" ::: "memory");
  }
  __syncthreads();   // Q (and Wsm) now visible block-wide

  // Per-lane base pointers (uniform trip count; EXEC stays all-ones for WMMA).
  const _Float16* Arow = Dh + ((size_t)b * NPIX + (m0 + (lane & 15))) * NPIX
                            + ((lane >> 4) << 3);      // +0 or +8 halves
  const _Float16* BloL = Qlds + (lane & 15) * QROW + ((lane >> 4) << 3);
  const _Float16* BhiL = BloL + 16 * QROW;             // classes 16..31

  v8f clo = {};
  v8f chi = {};

  for (int k = 0; k < NPIX; k += 32) {
    const _Float16* ap = Arow + k;
    v8h a0 = *(const v8h*)(ap);
    v8h a1 = *(const v8h*)(ap + 16);
    v16h A = __builtin_shufflevector(a0, a1, 0, 1, 2, 3, 4, 5, 6, 7,
                                             8, 9, 10, 11, 12, 13, 14, 15);

    v8h bl0 = *(const v8h*)(BloL + k);
    v8h bl1 = *(const v8h*)(BloL + k + 16);
    v16h BL = __builtin_shufflevector(bl0, bl1, 0, 1, 2, 3, 4, 5, 6, 7,
                                                8, 9, 10, 11, 12, 13, 14, 15);

    v8h bh0 = *(const v8h*)(BhiL + k);
    v8h bh1 = *(const v8h*)(BhiL + k + 16);
    v16h BH = __builtin_shufflevector(bh0, bh1, 0, 1, 2, 3, 4, 5, 6, 7,
                                                8, 9, 10, 11, 12, 13, 14, 15);

    __builtin_prefetch(ap + 256, 0, 3);  // global_prefetch_b8: D stream ahead

    clo = __builtin_amdgcn_wmma_f32_16x16x32_f16(false, A, false, BL,
                                                 (short)0, clo, false, false);
    chi = __builtin_amdgcn_wmma_f32_16x16x32_f16(false, A, false, BH,
                                                 (short)0, chi, false, false);
  }

  // C/D layout: lane j<16 -> pixels m0+0..7 in vgpr v, lane j>=16 -> m0+8..15;
  // column = (lane&15) for clo, 16+(lane&15) for chi.
  const int phalf = (lane >> 4) << 3;
  const int cls   = lane & 15;
#pragma unroll
  for (int vv = 0; vv < 8; ++vv) {
    Cs[wave][phalf + vv][cls]      = clo[vv];
    Cs[wave][phalf + vv][cls + 16] = chi[vv];
  }
  __syncthreads();   // publish Cs (cross-lane)

  if (lane < 16) {
    const int    p    = lane;
    const size_t base = (size_t)b * NCLS * NPIX + (size_t)(m0 + p);
    for (int d = 0; d < NCLS; ++d) {
      float acc = 0.f;
#pragma unroll
      for (int e = 0; e < NCLS; ++e) acc += Wsm[d * NCLS + e] * Cs[wave][p][e];
      out[base + (size_t)d * NPIX] = seg[base + (size_t)d * NPIX] - acc;
    }
  }
}

// ---------------------------------------------------------------------------
extern "C" void kernel_launch(void* const* d_in, const int* in_sizes, int n_in,
                              void* d_out, int out_size, void* d_ws, size_t ws_size,
                              hipStream_t stream) {
  (void)in_sizes; (void)n_in; (void)out_size; (void)ws_size;

  const float* seg = (const float*)d_in[0];   // [4,21,64,64] f32
  const float* W   = (const float*)d_in[1];   // [4,4096,4096] f32
  const float* wts = (const float*)d_in[2];   // [21,21] f32
  float*       out = (float*)d_out;           // [4,21,64,64] f32 (= qVals)

  _Float16* Dh = (_Float16*)d_ws;                                   // 128 MiB
  _Float16* Qh = (_Float16*)((char*)d_ws +
                             (size_t)BS * NPIX * NPIX * sizeof(_Float16));

  // Row-normalize W -> f16 D (one HBM pass; D then lives in the 192MB L2).
  k_rownorm<<<dim3(BS * NPIX), dim3(256), 0, stream>>>(W, Dh);

  for (int it = 0; it < NITER; ++it) {
    const float* qsrc = (it == 0) ? seg : out;   // qVals starts as seg
    k_softmax<<<dim3(BS * NPIX / 256), dim3(256), 0, stream>>>(qsrc, Qh);
    k_crf_step<<<dim3(BS * 32), dim3(256), QLDS_BYTES, stream>>>(Dh, Qh, seg,
                                                                 wts, out);
  }
}
